// mergeblock_18485539242519
// MI455X (gfx1250) — matrix operand order repaired and checked by reading, not compiled, not run on hardware
//
#include <hip/hip_runtime.h>

typedef __attribute__((ext_vector_type(16))) __bf16 v16bf;
typedef __attribute__((ext_vector_type(8)))  __bf16 v8bf;
typedef __attribute__((ext_vector_type(8)))  float  v8f;

#define Hh 256
#define Ww 256
#define HW 65536
#define Cc 64
#define Ss 16
#define Kdim 1152      // 128 channels * 9 taps
#define TP 130         // tile pixels (128 + halo)
#define CPAD 136       // 128 channels padded to 136 (16B-aligned fragments)

// ---------------------------------------------------------------------------
// conv1: sub = conv3x3(concat(x,bridge), W_sub)  -> Nbuf [B,16,HW] f32
// Implicit GEMM, M=16, N=16 px/wave, K=1152 (36 bf16 WMMA steps).
// Weights + pre-zeroed bf16 pixel tile staged in LDS; inner loop is
// 4x ds_load_b128 + 1x v_wmma per step, no predication.
// ---------------------------------------------------------------------------
__global__ __launch_bounds__(256) void conv1_wmma(
    const float* __restrict__ x, const float* __restrict__ bridge,
    const float* __restrict__ Wsub, float* __restrict__ Nbuf)
{
    __shared__ __bf16 Wlds[Ss * Kdim];          // 36 KB
    __shared__ __bf16 Bt[3 * TP * CPAD];        // 106 KB
    const int tid = threadIdx.x;
    const int b  = blockIdx.z;
    const int y  = blockIdx.y;
    const int xb0 = blockIdx.x * 128;

    const float* xb = x      + (size_t)b * Cc * HW;
    const float* bb = bridge + (size_t)b * Cc * HW;

    // stage weights: Wlds[s][K], K = tap*128 + c
    for (int idx = tid; idx < Ss * Kdim; idx += 256) {
        int s = idx / Kdim, K = idx % Kdim;
        int tap = K / 128, c = K % 128;
        Wlds[idx] = (__bf16)Wsub[(s * 128 + c) * 9 + tap];
    }
    // stage pixel tile: Bt[row][p][c], zero-filled halo (clamp + select, no branches)
    for (int idx = tid; idx < 3 * TP * 128; idx += 256) {
        int p   = idx % TP;
        int rem = idx / TP;
        int c   = rem % 128;
        int row = rem / 128;
        int yy = y + row - 1;
        int xx = xb0 + p - 1;
        bool inb = ((unsigned)yy < Hh) && ((unsigned)xx < Ww);
        int yyc = inb ? yy : 0;
        int xxc = inb ? xx : 0;
        size_t po = (size_t)yyc * Ww + xxc;
        float v = (c < 64) ? xb[(size_t)c * HW + po]
                           : bb[(size_t)(c - 64) * HW + po];
        Bt[(row * TP + p) * CPAD + c] = inb ? (__bf16)v : (__bf16)0.f;
    }
    __syncthreads();

    const int wave = tid >> 5;
    const int lane = tid & 31;
    const int ncol = lane & 15;
    const int half = lane >> 4;
    const int px    = xb0 + wave * 16 + ncol;   // global pixel x
    const int pbase = wave * 16 + ncol;         // local pixel (0..127)

    const __bf16* Arow = &Wlds[(lane & 15) * Kdim];

    v8f acc = {};
    for (int step = 0; step < 36; ++step) {
        const int tap = step >> 2, cg = step & 3;
        const int row = tap / 3, tx = tap % 3;

        const __bf16* ap = Arow + step * 32 + half * 16;
        v8bf a0 = *(const v8bf*)ap;
        v8bf a1 = *(const v8bf*)(ap + 8);
        v16bf aF = __builtin_shufflevector(a0, a1,
            0,1,2,3,4,5,6,7,8,9,10,11,12,13,14,15);

        const __bf16* bq = &Bt[(row * TP + pbase + tx) * CPAD + cg * 32 + half * 16];
        v8bf b0 = *(const v8bf*)bq;
        v8bf b1 = *(const v8bf*)(bq + 8);
        v16bf bF = __builtin_shufflevector(b0, b1,
            0,1,2,3,4,5,6,7,8,9,10,11,12,13,14,15);

        acc = __builtin_amdgcn_wmma_f32_16x16x32_bf16(
            false, aF, false, bF, (short)0, acc, false, false);
    }

    #pragma unroll
    for (int v = 0; v < 8; ++v) {
        int M = v + 8 * half;
        Nbuf[((size_t)b * Ss + M) * HW + (size_t)y * Ww + px] = acc[v];
    }
}

// ---------------------------------------------------------------------------
// zero G/P accumulators
// ---------------------------------------------------------------------------
__global__ void zero_gp(float* __restrict__ G, float* __restrict__ P)
{
    int i = blockIdx.x * 256 + threadIdx.x;
    if (i < 8 * 256)  G[i] = 0.f;
    if (i < 8 * 1024) P[i] = 0.f;
}

// ---------------------------------------------------------------------------
// G[b,s,t] = sum_n N[s,n]N[t,n]   P[b,s,c] = sum_n N[s,n]*bridge[c,n]
// ---------------------------------------------------------------------------
__global__ __launch_bounds__(256) void reduce_gp(
    const float* __restrict__ Nbuf, const float* __restrict__ bridge,
    float* __restrict__ G, float* __restrict__ P)
{
    __shared__ float Nt[16][128];
    __shared__ float Bt[64][128];
    const int tid = threadIdx.x;
    const int b = blockIdx.y;
    const int n0base = blockIdx.x * (HW / 16);

    const float* Nb = Nbuf   + (size_t)b * Ss * HW;
    const float* Br = bridge + (size_t)b * Cc * HW;

    const int gs = tid >> 4, gt = tid & 15;
    const int pc = tid & 63, ps0 = tid >> 6;

    float gAcc = 0.f;
    float pAcc[4] = {0.f, 0.f, 0.f, 0.f};

    for (int t0 = 0; t0 < 4096; t0 += 128) {
        const int n0 = n0base + t0;
        for (int idx = tid; idx < 2048; idx += 256) {
            int s = idx >> 7, j = idx & 127;
            Nt[s][j] = Nb[(size_t)s * HW + n0 + j];
        }
        for (int idx = tid; idx < 8192; idx += 256) {
            int c = idx >> 7, j = idx & 127;
            Bt[c][j] = Br[(size_t)c * HW + n0 + j];
        }
        __syncthreads();
        for (int j = 0; j < 128; ++j) {
            float ns = Nt[gs][j];
            gAcc += ns * Nt[gt][j];
            float bv = Bt[pc][j];
            pAcc[0] += Nt[ps0     ][j] * bv;
            pAcc[1] += Nt[ps0 +  4][j] * bv;
            pAcc[2] += Nt[ps0 +  8][j] * bv;
            pAcc[3] += Nt[ps0 + 12][j] * bv;
        }
        __syncthreads();
    }
    atomicAdd(&G[b * 256 + tid], gAcc);
    #pragma unroll
    for (int k = 0; k < 4; ++k)
        atomicAdd(&P[b * 1024 + (ps0 + 4 * k) * 64 + pc], pAcc[k]);
}

// ---------------------------------------------------------------------------
// Q = G^{-1} P  (16x16 Gauss-Jordan w/ partial pivot, then 16x64 matmul)
// ---------------------------------------------------------------------------
__global__ __launch_bounds__(64) void solve_q(
    const float* __restrict__ G, const float* __restrict__ P,
    float* __restrict__ Q)
{
    __shared__ float A[16][33];
    __shared__ float Pl[16][64];
    const int b = blockIdx.x;
    const int tid = threadIdx.x;

    for (int idx = tid; idx < 256; idx += 64) {
        int r = idx >> 4, c = idx & 15;
        A[r][c]      = G[b * 256 + idx];
        A[r][16 + c] = (r == c) ? 1.f : 0.f;
    }
    for (int idx = tid; idx < 1024; idx += 64)
        Pl[idx >> 6][idx & 63] = P[b * 1024 + idx];
    __syncthreads();

    if (tid == 0) {
        for (int k = 0; k < 16; ++k) {
            int piv = k; float best = fabsf(A[k][k]);
            for (int r = k + 1; r < 16; ++r) {
                float v = fabsf(A[r][k]);
                if (v > best) { best = v; piv = r; }
            }
            if (piv != k)
                for (int c = 0; c < 32; ++c) {
                    float t = A[k][c]; A[k][c] = A[piv][c]; A[piv][c] = t;
                }
            float d = A[k][k];
            float inv = (fabsf(d) > 1e-30f) ? 1.f / d : 0.f;
            for (int c = 0; c < 32; ++c) A[k][c] *= inv;
            for (int r = 0; r < 16; ++r) {
                if (r == k) continue;
                float f = A[r][k];
                for (int c = 0; c < 32; ++c) A[r][c] -= f * A[k][c];
            }
        }
    }
    __syncthreads();

    for (int idx = tid; idx < 1024; idx += 64) {
        int s = idx >> 6, c = idx & 63;
        float acc = 0.f;
        for (int t = 0; t < 16; ++t) acc += A[s][16 + t] * Pl[t][c];
        Q[b * 1024 + idx] = acc;
    }
}

// ---------------------------------------------------------------------------
// bridge_proj[b,c,n] = sum_s N[s,n] * Q[s,c]  -> bf16 buffer for conv2
// ---------------------------------------------------------------------------
__global__ __launch_bounds__(256) void proj_bp(
    const float* __restrict__ Nbuf, const float* __restrict__ Q,
    __bf16* __restrict__ bp)
{
    __shared__ float Ql[16][64];
    const int b = blockIdx.y;
    const int tid = threadIdx.x;
    for (int idx = tid; idx < 1024; idx += 256)
        Ql[idx >> 6][idx & 63] = Q[b * 1024 + idx];
    __syncthreads();

    const int n = blockIdx.x * 256 + tid;
    const float* Nb = Nbuf + (size_t)b * Ss * HW;
    float nv[16];
    #pragma unroll
    for (int s = 0; s < 16; ++s) nv[s] = Nb[(size_t)s * HW + n];

    __bf16* bpb = bp + (size_t)b * Cc * HW;
    for (int c = 0; c < 64; ++c) {
        float acc = 0.f;
        #pragma unroll
        for (int s = 0; s < 16; ++s) acc += nv[s] * Ql[s][c];
        bpb[(size_t)c * HW + n] = (__bf16)acc;
    }
}

// ---------------------------------------------------------------------------
// conv2: out = conv3x3(concat(x, bridge_proj), W_cb) + x
// Same structure as conv1; blockIdx.z = b*4 + mtile (16 out ch per block)
// ---------------------------------------------------------------------------
__global__ __launch_bounds__(256) void conv2_wmma(
    const float* __restrict__ x, const __bf16* __restrict__ bpg,
    const float* __restrict__ Wcb, float* __restrict__ out)
{
    __shared__ __bf16 Wlds[Ss * Kdim];          // 36 KB
    __shared__ __bf16 Bt[3 * TP * CPAD];        // 106 KB
    const int tid = threadIdx.x;
    const int bz = blockIdx.z;
    const int b = bz >> 2, m0 = (bz & 3) * 16;
    const int y  = blockIdx.y;
    const int xb0 = blockIdx.x * 128;

    const float*  xb  = x   + (size_t)b * Cc * HW;
    const __bf16* bpb = bpg + (size_t)b * Cc * HW;

    for (int idx = tid; idx < Ss * Kdim; idx += 256) {
        int m = idx / Kdim, K = idx % Kdim;
        int tap = K / 128, c = K % 128;
        Wlds[idx] = (__bf16)Wcb[((m0 + m) * 128 + c) * 9 + tap];
    }
    for (int idx = tid; idx < 3 * TP * 128; idx += 256) {
        int p   = idx % TP;
        int rem = idx / TP;
        int c   = rem % 128;
        int row = rem / 128;
        int yy = y + row - 1;
        int xx = xb0 + p - 1;
        bool inb = ((unsigned)yy < Hh) && ((unsigned)xx < Ww);
        int yyc = inb ? yy : 0;
        int xxc = inb ? xx : 0;
        size_t po = (size_t)yyc * Ww + xxc;
        __bf16 v;
        if (c < 64) v = (__bf16)xb[(size_t)c * HW + po];
        else        v = bpb[(size_t)(c - 64) * HW + po];
        Bt[(row * TP + p) * CPAD + c] = inb ? v : (__bf16)0.f;
    }
    __syncthreads();

    const int wave = tid >> 5;
    const int lane = tid & 31;
    const int ncol = lane & 15;
    const int half = lane >> 4;
    const int px    = xb0 + wave * 16 + ncol;
    const int pbase = wave * 16 + ncol;

    const __bf16* Arow = &Wlds[(lane & 15) * Kdim];

    v8f acc = {};
    for (int step = 0; step < 36; ++step) {
        const int tap = step >> 2, cg = step & 3;
        const int row = tap / 3, tx = tap % 3;

        const __bf16* ap = Arow + step * 32 + half * 16;
        v8bf a0 = *(const v8bf*)ap;
        v8bf a1 = *(const v8bf*)(ap + 8);
        v16bf aF = __builtin_shufflevector(a0, a1,
            0,1,2,3,4,5,6,7,8,9,10,11,12,13,14,15);

        const __bf16* bq = &Bt[(row * TP + pbase + tx) * CPAD + cg * 32 + half * 16];
        v8bf b0 = *(const v8bf*)bq;
        v8bf b1 = *(const v8bf*)(bq + 8);
        v16bf bF = __builtin_shufflevector(b0, b1,
            0,1,2,3,4,5,6,7,8,9,10,11,12,13,14,15);

        acc = __builtin_amdgcn_wmma_f32_16x16x32_bf16(
            false, aF, false, bF, (short)0, acc, false, false);
    }

    #pragma unroll
    for (int v = 0; v < 8; ++v) {
        int M = m0 + v + 8 * half;
        size_t off = ((size_t)b * Cc + M) * HW + (size_t)y * Ww + px;
        out[off] = acc[v] + xb[(size_t)M * HW + (size_t)y * Ww + px];
    }
}

// ---------------------------------------------------------------------------
// launch
// ---------------------------------------------------------------------------
extern "C" void kernel_launch(void* const* d_in, const int* in_sizes, int n_in,
                              void* d_out, int out_size, void* d_ws, size_t ws_size,
                              hipStream_t stream)
{
    const float* x      = (const float*)d_in[0];
    const float* bridge = (const float*)d_in[1];
    const float* Wsub   = (const float*)d_in[2];
    const float* Wcb    = (const float*)d_in[3];
    float* out          = (float*)d_out;

    char* ws = (char*)d_ws;
    size_t off = 0;
    float* Nbuf = (float*)(ws + off); off += (size_t)8 * Ss * HW * sizeof(float);
    float* Gb   = (float*)(ws + off); off += (size_t)8 * 256 * sizeof(float);
    float* Pb   = (float*)(ws + off); off += (size_t)8 * 1024 * sizeof(float);
    float* Qb   = (float*)(ws + off); off += (size_t)8 * 1024 * sizeof(float);
    __bf16* bp  = (__bf16*)(ws + off);

    zero_gp   <<<32, 256, 0, stream>>>(Gb, Pb);
    conv1_wmma<<<dim3(Ww / 128, Hh, 8),  256, 0, stream>>>(x, bridge, Wsub, Nbuf);
    reduce_gp <<<dim3(16, 8),            256, 0, stream>>>(Nbuf, bridge, Gb, Pb);
    solve_q   <<<8, 64, 0, stream>>>(Gb, Pb, Qb);
    proj_bp   <<<dim3(HW / 256, 8),      256, 0, stream>>>(Nbuf, Qb, bp);
    conv2_wmma<<<dim3(Ww / 128, Hh, 32), 256, 0, stream>>>(x, bp, Wcb, out);
}
